// Attention_54047868453397
// MI455X (gfx1250) — compile-verified, matrix operand order checked by reading, not run on hardware
//
#include <hip/hip_runtime.h>
#include <hip/hip_bf16.h>

#define B_ 4
#define N_ 2048
#define DIM_ 1024
#define HEADS_ 16
#define DH_ 64
#define INNER_ 1024
#define NP_ 2080              // 1 null + 2048 keys + 31 pad (multiple of 32)
#define M_ (B_*N_)
#define EPS_ 1e-5f

typedef __attribute__((ext_vector_type(16))) __bf16 v16bf;
typedef __attribute__((ext_vector_type(8)))  float  v8f;
typedef __attribute__((ext_vector_type(4)))  float  f4;   // trivial, union-safe
typedef __attribute__((ext_vector_type(4)))  int    i4;

union Frag  { v16bf v; f4 f[2]; unsigned short us[16]; };
union Pack8 { f4 f; unsigned short us[8]; };

// ---- feature detection: gfx1250 async global->LDS path -----------------
#if defined(__has_builtin)
#if __has_builtin(__builtin_amdgcn_global_load_async_to_lds_b128) && \
    __has_builtin(__builtin_amdgcn_s_wait_asynccnt)
#define HAS_ASYNC_LDS 1
#endif
#endif
#ifndef HAS_ASYNC_LDS
#define HAS_ASYNC_LDS 0
#endif

#if HAS_ASYNC_LDS
typedef __attribute__((address_space(1))) i4 gi4;   // global int4
typedef __attribute__((address_space(3))) i4 li4;   // LDS int4
#endif

__device__ inline void ld_async_b128(const unsigned short* g, unsigned short* l) {
#if HAS_ASYNC_LDS
  __builtin_amdgcn_global_load_async_to_lds_b128((gi4*)g, (li4*)l, 0, 0);
#else
  *(f4*)l = *(const f4*)g;
#endif
}
__device__ inline void wait_async() {
#if HAS_ASYNC_LDS
  __builtin_amdgcn_s_wait_asynccnt(0);
#endif
}
__device__ inline void wait_ds() {
#if defined(__has_builtin)
#if __has_builtin(__builtin_amdgcn_s_wait_dscnt)
  __builtin_amdgcn_s_wait_dscnt(0);
#endif
#endif
}

__device__ inline unsigned short f2bf(float x) {
  unsigned int u = __float_as_uint(x);
  u += 0x7fffu + ((u >> 16) & 1u);          // round-to-nearest-even
  return (unsigned short)(u >> 16);
}

__device__ inline v8f wmma_bf16(v16bf a, v16bf b, v8f c) {
  return __builtin_amdgcn_wmma_f32_16x16x32_bf16(false, a, false, b, (short)0, c, false, false);
}

// Fragment fill matching the 16-bit A/B VGPR layout:
// lane L: row = L&15, kbase = (L<16)?0:8; elems kbase..kbase+7 and kbase+16..kbase+23
__device__ inline v16bf load_frag(const unsigned short* rowbase, int kbase) {
  Frag fr;
  fr.f[0] = *(const f4*)(rowbase + kbase);
  fr.f[1] = *(const f4*)(rowbase + kbase + 16);
  return fr.v;
}

// ---------------------------------------------------------------- utilities
__global__ void cvt_bf16(const float* __restrict__ in, unsigned short* __restrict__ out, int n) {
  int i = blockIdx.x * blockDim.x + threadIdx.x;
  if (i < n) out[i] = f2bf(in[i]);
}

__global__ __launch_bounds__(256) void layernorm_k(
    const float* __restrict__ x, const float* __restrict__ gamma,
    unsigned short* __restrict__ xn) {
  __shared__ float red[2][8];
  const int row = blockIdx.x, tid = threadIdx.x;
  const float* xr = x + (size_t)row * DIM_;
  f4 v = *(const f4*)(xr + tid * 4);
  float s1 = v.x + v.y + v.z + v.w;
  float s2 = v.x * v.x + v.y * v.y + v.z * v.z + v.w * v.w;
  #pragma unroll
  for (int d = 1; d <= 16; d <<= 1) { s1 += __shfl_xor(s1, d, 32); s2 += __shfl_xor(s2, d, 32); }
  const int wave = tid >> 5, lane = tid & 31;
  if (lane == 0) { red[0][wave] = s1; red[1][wave] = s2; }
  __syncthreads();
  float t1 = 0.f, t2 = 0.f;
  #pragma unroll
  for (int i = 0; i < 8; ++i) { t1 += red[0][i]; t2 += red[1][i]; }
  const float mean = t1 * (1.0f / DIM_);
  const float var  = t2 * (1.0f / DIM_) - mean * mean;
  const float rstd = rsqrtf(var + EPS_);
  f4 g = *(const f4*)(gamma + tid * 4);
  unsigned long long pk =
      (unsigned long long)f2bf((v.x - mean) * rstd * g.x)
    | ((unsigned long long)f2bf((v.y - mean) * rstd * g.y) << 16)
    | ((unsigned long long)f2bf((v.z - mean) * rstd * g.z) << 32)
    | ((unsigned long long)f2bf((v.w - mean) * rstd * g.w) << 48);
  *(unsigned long long*)(xn + (size_t)row * DIM_ + tid * 4) = pk;
}

// ---------------------------------------------------------------- WMMA GEMM
// 128x128 block tile, 8 waves in 4x2, each wave 32x64 (8 WMMAs / k-step).
//  mode 0: l2norm(head-row)*q_scale*DH^-0.5 -> qbuf[b,h,n,dh]
//  mode 1: l2norm(head-row)*k_scale         -> kbuf[b,h,n+1,dh] (null slot 0)
//  mode 2: plain                            -> vbuf[b,h,n+1,dh]
//  mode 3: plain f32                        -> out[M x 1024]
__global__ __launch_bounds__(256) void gemm_proj(
    const unsigned short* __restrict__ A,
    const unsigned short* __restrict__ Bw,
    int Kdim, int ldb, int col0, int mode,
    const float* __restrict__ svec,
    unsigned short* __restrict__ out_bf,
    float* __restrict__ out_f32) {
  __shared__ __align__(16) unsigned short lds_a[128 * 40];    // row-major, stride 40
  __shared__ __align__(16) unsigned short lds_bT[128 * 40];   // transposed (n-major)
  const int tid = threadIdx.x;
  const int wave = tid >> 5, lane = tid & 31;
  const int n16 = lane & 15;
  const int kbase = (lane < 16) ? 0 : 8;
  const int mbase = (lane < 16) ? 0 : 8;
  const int wy = wave >> 1, wx = wave & 1;
  const int m0 = blockIdx.x * 128;
  const int n0 = blockIdx.y * 128 + col0;

  v8f acc[2][4] = {};
  for (int kk = 0; kk < Kdim; kk += 32) {
    #pragma unroll
    for (int i = 0; i < 2; ++i) {                 // A tile 128x32 (async -> LDS)
      int lin = (tid * 2 + i) * 8;
      int r = lin >> 5, c = lin & 31;
      ld_async_b128(A + (size_t)(m0 + r) * Kdim + kk + c, &lds_a[r * 40 + c]);
    }
    #pragma unroll
    for (int i = 0; i < 2; ++i) {                 // B tile 32x128, store transposed
      int lin = (tid * 2 + i) * 8;
      int kr = lin >> 7, c = lin & 127;
      Pack8 p; p.f = *(const f4*)(Bw + (size_t)(kk + kr) * ldb + n0 + c);
      #pragma unroll
      for (int e = 0; e < 8; ++e) lds_bT[(c + e) * 40 + kr] = p.us[e];
    }
    wait_async();
    __syncthreads();
    v16bf af[2], bfr[4];
    af[0] = load_frag(&lds_a[(wy * 32 + n16) * 40], kbase);
    af[1] = load_frag(&lds_a[(wy * 32 + 16 + n16) * 40], kbase);
    #pragma unroll
    for (int t = 0; t < 4; ++t)
      bfr[t] = load_frag(&lds_bT[(wx * 64 + t * 16 + n16) * 40], kbase);
    #pragma unroll
    for (int i = 0; i < 2; ++i)
      #pragma unroll
      for (int t = 0; t < 4; ++t)
        acc[i][t] = wmma_bf16(af[i], bfr[t], acc[i][t]);
    __syncthreads();
  }

  const int head = blockIdx.y * 2 + wx;           // modes 0..2 (Nout=1024, grid.y=8)
  float sv[4];
  #pragma unroll
  for (int t = 0; t < 4; ++t)
    sv[t] = (mode <= 1) ? svec[t * 16 + n16] * (mode == 0 ? 0.125f : 1.0f) : 1.0f;

  #pragma unroll
  for (int i = 0; i < 2; ++i) {
    #pragma unroll
    for (int j = 0; j < 8; ++j) {
      float rn = 1.0f;
      if (mode <= 1) {
        float ss = 0.f;
        #pragma unroll
        for (int t = 0; t < 4; ++t) ss += acc[i][t][j] * acc[i][t][j];
        #pragma unroll
        for (int d = 1; d <= 8; d <<= 1) ss += __shfl_xor(ss, d, 32);
        rn = rsqrtf(ss + 1e-12f);
      }
      const int gm = m0 + wy * 32 + i * 16 + mbase + j;
      const int b  = gm >> 11;           // / 2048 (modes 0..2)
      const int nn = gm & 2047;
      #pragma unroll
      for (int t = 0; t < 4; ++t) {
        float val = acc[i][t][j];
        const int col = t * 16 + n16;
        if (mode == 0) {
          out_bf[(((size_t)(b * HEADS_ + head) * N_) + nn) * DH_ + col] = f2bf(val * rn * sv[t]);
        } else if (mode == 1) {
          out_bf[(((size_t)(b * HEADS_ + head) * NP_) + nn + 1) * DH_ + col] = f2bf(val * rn * sv[t]);
        } else if (mode == 2) {
          out_bf[(((size_t)(b * HEADS_ + head) * NP_) + nn + 1) * DH_ + col] = f2bf(val);
        } else {
          out_f32[(size_t)gm * DIM_ + (size_t)blockIdx.y * 128 + wx * 64 + col] = val;
        }
      }
    }
  }
}

// ---------------------------------------------------------------- null token / pad / bias
__global__ __launch_bounds__(64) void fill_null_pad(
    const float* __restrict__ null_kv, const float* __restrict__ k_scale,
    unsigned short* __restrict__ kbuf, unsigned short* __restrict__ vbuf) {
  const int blk = blockIdx.x, b = blk >> 4, h = blk & 15, d = threadIdx.x;
  const float* nk = null_kv + (size_t)h * DH_;
  const float* nv = null_kv + (size_t)(HEADS_ + h) * DH_;
  float ss = 1e-12f;
  for (int i = 0; i < DH_; ++i) { float t = nk[i]; ss += t * t; }
  const float rn = rsqrtf(ss);
  const size_t base = ((size_t)(b * HEADS_ + h) * NP_) * DH_;
  kbuf[base + d] = f2bf(nk[d] * rn * k_scale[d]);
  vbuf[base + d] = f2bf(nv[d]);
  for (int r = N_ + 1; r < NP_; ++r) {
    kbuf[base + (size_t)r * DH_ + d] = 0;
    vbuf[base + (size_t)r * DH_ + d] = 0;
  }
}

__global__ void fill_bias(const int* __restrict__ mask, float* __restrict__ bias) {
  int i = blockIdx.x * blockDim.x + threadIdx.x;
  if (i >= B_ * NP_) return;
  const int b = i / NP_, j = i % NP_;
  float v;
  if (j == 0)       v = 0.f;
  else if (j <= N_) v = mask[b * N_ + j - 1] ? 0.f : -10000.f;
  else              v = -30000.f;
  bias[i] = v;
}

// ---------------------------------------------------------------- flash attention
__global__ __launch_bounds__(256) void attn_kernel(
    const unsigned short* __restrict__ qb,
    const unsigned short* __restrict__ kb,
    const unsigned short* __restrict__ vb,
    const float* __restrict__ bias,
    unsigned short* __restrict__ attnout) {
  __shared__ __align__(16) unsigned short lds_k[32 * 72];       // keys x dh (async target)
  __shared__ __align__(16) unsigned short lds_vT[64 * 40];      // dh x keys
  __shared__ __align__(16) unsigned short lds_p[8 * 16 * 40];   // per-wave P scratch
  const int tid = threadIdx.x;
  const int wave = tid >> 5, lane = tid & 31;
  const int n16 = lane & 15;
  const int kbase = (lane < 16) ? 0 : 8;
  const int mbase = (lane < 16) ? 0 : 8;
  const int bh = blockIdx.y;
  const int b = bh >> 4, h = bh & 15;
  const int q0 = blockIdx.x * 128 + wave * 16;

  const unsigned short* qrow = qb + ((size_t)bh * N_ + q0 + n16) * DH_;
  v16bf aq0 = load_frag(qrow, kbase);          // Q dh 0..31
  v16bf aq1 = load_frag(qrow + 32, kbase);     // Q dh 32..63

  float m_i[8], l_i[8];
  v8f acc[4] = {};
  #pragma unroll
  for (int j = 0; j < 8; ++j) { m_i[j] = -1e30f; l_i[j] = 0.f; }

  unsigned short* wp = &lds_p[wave * 16 * 40];

  for (int kc = 0; kc < NP_; kc += 32) {
    {
      int lin = tid * 8;
      int kr = lin >> 6, c = lin & 63;
      const size_t kvrow = ((size_t)bh * NP_ + kc + kr) * DH_ + c;
      ld_async_b128(kb + kvrow, &lds_k[kr * 72 + c]);      // K: async -> LDS
      Pack8 pv; pv.f = *(const f4*)(vb + kvrow);           // V: transpose by hand
      #pragma unroll
      for (int e = 0; e < 8; ++e) lds_vT[(c + e) * 40 + kr] = pv.us[e];
    }
    wait_async();
    __syncthreads();

    const float bias0 = bias[b * NP_ + kc + n16];
    const float bias1 = bias[b * NP_ + kc + 16 + n16];

    v16bf kf[4];
    kf[0] = load_frag(&lds_k[n16 * 72], kbase);
    kf[1] = load_frag(&lds_k[n16 * 72] + 32, kbase);
    kf[2] = load_frag(&lds_k[(16 + n16) * 72], kbase);
    kf[3] = load_frag(&lds_k[(16 + n16) * 72] + 32, kbase);
    v8f S0 = {}, S1 = {};
    S0 = wmma_bf16(aq0, kf[0], S0);
    S0 = wmma_bf16(aq1, kf[1], S0);
    S1 = wmma_bf16(aq0, kf[2], S1);
    S1 = wmma_bf16(aq1, kf[3], S1);

    #pragma unroll
    for (int j = 0; j < 8; ++j) {
      float s0 = S0[j] + bias0;
      float s1 = S1[j] + bias1;
      float mx = fmaxf(s0, s1);
      #pragma unroll
      for (int d = 1; d <= 8; d <<= 1) mx = fmaxf(mx, __shfl_xor(mx, d, 32));
      const float m_new = fmaxf(m_i[j], mx);
      const float alpha = __expf(m_i[j] - m_new);
      const float p0 = __expf(s0 - m_new);
      const float p1 = __expf(s1 - m_new);
      float rs = p0 + p1;
      #pragma unroll
      for (int d = 1; d <= 8; d <<= 1) rs += __shfl_xor(rs, d, 32);
      l_i[j] = l_i[j] * alpha + rs;
      m_i[j] = m_new;
      #pragma unroll
      for (int t = 0; t < 4; ++t) acc[t][j] *= alpha;
      wp[(mbase + j) * 40 + n16]      = f2bf(p0);
      wp[(mbase + j) * 40 + 16 + n16] = f2bf(p1);
    }
    wait_ds();   // same-wave LDS transpose round-trip (DS pipe is in-order per wave)
    v16bf pf = load_frag(&wp[n16 * 40], kbase);
    v16bf bv[4];
    #pragma unroll
    for (int t = 0; t < 4; ++t)
      bv[t] = load_frag(&lds_vT[(t * 16 + n16) * 40], kbase);
    #pragma unroll
    for (int t = 0; t < 4; ++t)
      acc[t] = wmma_bf16(pf, bv[t], acc[t]);
    __syncthreads();
  }

  #pragma unroll
  for (int j = 0; j < 8; ++j) {
    const float inv = 1.0f / l_i[j];
    const int row = b * N_ + q0 + mbase + j;
    #pragma unroll
    for (int t = 0; t < 4; ++t)
      attnout[(size_t)row * INNER_ + h * DH_ + t * 16 + n16] = f2bf(acc[t][j] * inv);
  }
}

// ---------------------------------------------------------------- launch
extern "C" void kernel_launch(void* const* d_in, const int* in_sizes, int n_in,
                              void* d_out, int out_size, void* d_ws, size_t ws_size,
                              hipStream_t stream) {
  const float* x       = (const float*)d_in[0];
  const int*   cmask   = (const int*)d_in[1];
  const float* gamma   = (const float*)d_in[2];
  const float* null_kv = (const float*)d_in[3];
  const float* Wq      = (const float*)d_in[4];
  const float* Wkv     = (const float*)d_in[5];
  const float* q_scale = (const float*)d_in[6];
  const float* k_scale = (const float*)d_in[7];
  const float* Wout    = (const float*)d_in[8];
  float* out = (float*)d_out;

  char* ws = (char*)d_ws;
  size_t off = 0;
  auto alloc = [&](size_t bytes) {
    size_t o = off; off += (bytes + 255) & ~(size_t)255; return o;
  };
  unsigned short* xnb   = (unsigned short*)(ws + alloc((size_t)M_ * DIM_ * 2));
  unsigned short* wqb   = (unsigned short*)(ws + alloc((size_t)DIM_ * INNER_ * 2));
  unsigned short* wkvb  = (unsigned short*)(ws + alloc((size_t)DIM_ * 2 * INNER_ * 2));
  unsigned short* woutb = (unsigned short*)(ws + alloc((size_t)INNER_ * DIM_ * 2));
  unsigned short* qbuf  = (unsigned short*)(ws + alloc((size_t)B_ * HEADS_ * N_ * DH_ * 2));
  unsigned short* kbuf  = (unsigned short*)(ws + alloc((size_t)B_ * HEADS_ * NP_ * DH_ * 2));
  unsigned short* vbuf  = (unsigned short*)(ws + alloc((size_t)B_ * HEADS_ * NP_ * DH_ * 2));
  unsigned short* attnb = (unsigned short*)(ws + alloc((size_t)M_ * INNER_ * 2));
  float*          biasb = (float*)(ws + alloc((size_t)B_ * NP_ * 4));
  (void)ws_size; (void)n_in; (void)in_sizes; (void)out_size;

  cvt_bf16<<<(DIM_ * INNER_ + 255) / 256, 256, 0, stream>>>(Wq, wqb, DIM_ * INNER_);
  cvt_bf16<<<(DIM_ * 2 * INNER_ + 255) / 256, 256, 0, stream>>>(Wkv, wkvb, DIM_ * 2 * INNER_);
  cvt_bf16<<<(INNER_ * DIM_ + 255) / 256, 256, 0, stream>>>(Wout, woutb, INNER_ * DIM_);

  layernorm_k<<<M_, 256, 0, stream>>>(x, gamma, xnb);

  gemm_proj<<<dim3(M_ / 128, 8), 256, 0, stream>>>(xnb, wqb,  DIM_, INNER_,     0,      0, q_scale, qbuf, nullptr);
  gemm_proj<<<dim3(M_ / 128, 8), 256, 0, stream>>>(xnb, wkvb, DIM_, 2 * INNER_, 0,      1, k_scale, kbuf, nullptr);
  gemm_proj<<<dim3(M_ / 128, 8), 256, 0, stream>>>(xnb, wkvb, DIM_, 2 * INNER_, INNER_, 2, nullptr, vbuf, nullptr);

  fill_null_pad<<<B_ * HEADS_, 64, 0, stream>>>(null_kv, k_scale, kbuf, vbuf);
  fill_bias<<<(B_ * NP_ + 255) / 256, 256, 0, stream>>>(cmask, biasb);

  attn_kernel<<<dim3(N_ / 128, B_ * HEADS_), 256, 0, stream>>>(qbuf, kbuf, vbuf, biasb, attnb);

  gemm_proj<<<dim3(M_ / 128, 8), 256, 0, stream>>>(attnb, woutb, INNER_, DIM_, 0, 3, nullptr, nullptr, out);
}
